// Token_QK_Attention_3848290697742
// MI455X (gfx1250) — matrix operand order repaired and checked by reading, not compiled, not run on hardware
//
#include <hip/hip_runtime.h>
#include <hip/hip_fp16.h>

typedef __attribute__((ext_vector_type(16))) _Float16 v16h;
typedef __attribute__((ext_vector_type(8)))  _Float16 v8h;
typedef __attribute__((ext_vector_type(8)))  float    v8f;

// Explicit global-address-space pointers: survive the inline-asm re-opaquing
// with provenance intact so weight fetches lower to global_load_b128
// (LOADcnt only) instead of flat_load_b128 (LOADcnt + DScnt coupling).
typedef const _Float16 __attribute__((address_space(1)))* gwp_t;
typedef const v8h      __attribute__((address_space(1)))* gv8_t;

namespace {
constexpr int T_STEPS   = 4;
constexpr int BATCH     = 32;
constexpr int CCH       = 384;   // channels
constexpr int NPIX      = 256;   // H*W
constexpr int NT        = 32;    // columns per workgroup
constexpr int NT_TILES  = NT / 16;          // 2
constexpr int KTILES    = CCH / 32;         // 12
constexpr int WAVES     = 8;
constexpr int MT_PER_WAVE = (CCH / 16) / WAVES;  // 3 m-tiles per wave
constexpr int HEADS_N   = 8;
constexpr int SSTR      = CCH + 8;          // LDS K stride (halfs), padded
constexpr float EPS     = 1e-5f;

union Frag16 { v16h v; v8h h[2]; };
}

// ---------------------------------------------------------------------------
// Prep: fold BN (eval) + optional conv bias into f16 weights / f32 bias.
//   w'[d][c] = w[d][c] * gamma[d]/sqrt(var[d]+eps)
//   b'[d]    = conv_b[d]*inv[d] + beta[d] - mean[d]*inv[d]
// ---------------------------------------------------------------------------
__global__ void __launch_bounds__(256)
prep_weights(const float* __restrict__ w, const float* __restrict__ gamma,
             const float* __restrict__ beta, const float* __restrict__ mean,
             const float* __restrict__ var, const float* __restrict__ conv_bias,
             _Float16* __restrict__ w_h, float* __restrict__ bias_out)
{
  const int i = blockIdx.x * 256 + threadIdx.x;
  if (i < CCH * CCH) {
    const int d = i / CCH;
    const float inv = gamma[d] * rsqrtf(var[d] + EPS);
    w_h[i] = (_Float16)(w[i] * inv);
  }
  if (i < CCH) {
    const float inv = gamma[i] * rsqrtf(var[i] + EPS);
    const float cb  = conv_bias ? conv_bias[i] : 0.0f;
    bias_out[i] = cb * inv + beta[i] - mean[i] * inv;
  }
}

// ---------------------------------------------------------------------------
// Fully fused SNN token-QK attention. One WG = one (batch, 32-column) slice,
// loops t sequentially with all recurrent LIF state resident in registers.
// Weights stream from L2 each timestep (pointer is re-opaqued per t so the
// compiler cannot hoist + spill the A-fragments across timesteps).
// ---------------------------------------------------------------------------
__global__ void __launch_bounds__(256, 1)
snn_token_qk_attn(const float* __restrict__ x,
                  const _Float16* __restrict__ wq,
                  const _Float16* __restrict__ wk,
                  const _Float16* __restrict__ wp,
                  const float* __restrict__ bq,
                  const float* __restrict__ bk,
                  const float* __restrict__ bp,
                  const float* __restrict__ alpha_p,
                  float* __restrict__ out)
{
  __shared__ _Float16 s_lds[NT][SSTR];     // input spikes, B-operand layout [n][k]
  __shared__ _Float16 y_lds[NT][SSTR];     // k spikes, B-operand layout [n][k]
  __shared__ float attn_sum[HEADS_N * NT];
  __shared__ float attn_spk[HEADS_N * NT];

  const int tid  = threadIdx.x;
  const int wave = tid >> 5;
  const int lane = tid & 31;
  const int g    = lane >> 4;   // lane group (K/M-half select)
  const int ln   = lane & 15;   // row (A) / column (B,C,D) within tile
  const int b    = blockIdx.x / (NPIX / NT);
  const int n0   = (blockIdx.x % (NPIX / NT)) * NT;
  const float alpha = alpha_p[0];
  const v8f vzero = {};

  // Folded-BN biases for the C/D rows this lane owns (d = mt*16 + 8*g + r)
  float biasq[MT_PER_WAVE][8], biask[MT_PER_WAVE][8], biasp[MT_PER_WAVE][8];
#pragma unroll
  for (int j = 0; j < MT_PER_WAVE; ++j) {
    const int mt = wave + WAVES * j;
#pragma unroll
    for (int r = 0; r < 8; ++r) {
      const int d = mt * 16 + g * 8 + r;
      biasq[j][r] = bq[d]; biask[j][r] = bk[d]; biasp[j][r] = bp[d];
    }
  }

  // Recurrent state (persists across the t loop)
  float v_in[48];
#pragma unroll
  for (int i = 0; i < 48; ++i) v_in[i] = 0.0f;
  v8f vq[MT_PER_WAVE][NT_TILES], vk[MT_PER_WAVE][NT_TILES];
  v8f mst[MT_PER_WAVE][NT_TILES];
#pragma unroll
  for (int j = 0; j < MT_PER_WAVE; ++j)
#pragma unroll
    for (int nt = 0; nt < NT_TILES; ++nt) {
      vq[j][nt] = vzero; vk[j][nt] = vzero; mst[j][nt] = vzero;
    }
  float v_attn = 0.0f;

  for (int t = 0; t < T_STEPS; ++t) {
    // Re-opaque the (global-AS) weight pointers: blocks LICM from hoisting
    // (and then spilling) t-invariant A-fragments across timesteps.
    gwp_t wq_t = (gwp_t)wq;
    gwp_t wk_t = (gwp_t)wk;
    gwp_t wp_t = (gwp_t)wp;
    asm volatile("" : "+s"(wq_t), "+s"(wk_t), "+s"(wp_t));

    // ---- stage 1: input LIF -> binary f16 spikes into LDS ----
    const float* xt = x + ((size_t)(t * BATCH + b) * CCH) * NPIX + n0;
#pragma unroll
    for (int i = 0; i < 48; ++i) {
      const int e = i * 256 + tid;
      const int c = e >> 5;          // channel
      const int n = e & 31;          // column (coalesced across lanes)
      const float v = 0.5f * v_in[i] + xt[(size_t)c * NPIX + n];
      const float s = (v >= 1.0f) ? 1.0f : 0.0f;
      v_in[i] = v * (1.0f - s);
      s_lds[n][c] = (_Float16)s;
    }
    attn_sum[tid] = 0.0f;
    __syncthreads();

    // ---- stage 2: Q = Wq*S and K = Wk*S via v_wmma_f32_16x16x32_f16 ----
    v8f qacc[MT_PER_WAVE][NT_TILES], kacc[MT_PER_WAVE][NT_TILES];
#pragma unroll
    for (int j = 0; j < MT_PER_WAVE; ++j)
#pragma unroll
      for (int nt = 0; nt < NT_TILES; ++nt) { qacc[j][nt] = vzero; kacc[j][nt] = vzero; }

#pragma unroll 1                     // rolled: bounds in-flight fragments
    for (int kt = 0; kt < KTILES; ++kt) {
      Frag16 bf[NT_TILES];                              // B frags shared by Q & K
#pragma unroll
      for (int nt = 0; nt < NT_TILES; ++nt) {
        const _Float16* sp = &s_lds[nt * 16 + ln][kt * 32 + g * 16];
        bf[nt].h[0] = *(const v8h*)sp;
        bf[nt].h[1] = *(const v8h*)(sp + 8);
      }
#pragma unroll
      for (int j = 0; j < MT_PER_WAVE; ++j) {
        const int mt = wave + WAVES * j;
        const size_t row = (size_t)(mt * 16 + ln) * CCH + kt * 32 + g * 16;
        Frag16 aq, ak;                                  // A frags straight from L2
        aq.h[0] = *(gv8_t)(wq_t + row);
        aq.h[1] = *(gv8_t)(wq_t + row + 8);
        ak.h[0] = *(gv8_t)(wk_t + row);
        ak.h[1] = *(gv8_t)(wk_t + row + 8);
#pragma unroll
        for (int nt = 0; nt < NT_TILES; ++nt) {
          qacc[j][nt] = __builtin_amdgcn_wmma_f32_16x16x32_f16(
              false, aq.v, false, bf[nt].v, (short)0, qacc[j][nt], false, false);
          kacc[j][nt] = __builtin_amdgcn_wmma_f32_16x16x32_f16(
              false, ak.v, false, bf[nt].v, (short)0, kacc[j][nt], false, false);
        }
      }
    }

    // ---- stage 3: bias + LIF on q,k; k-spikes -> y_lds; memory scan; partials ----
#pragma unroll
    for (int j = 0; j < MT_PER_WAVE; ++j) {
      const int mt = wave + WAVES * j;
      const int h  = mt / 3;                 // 48 rows/head == 3 m-tiles, aligned
#pragma unroll
      for (int nt = 0; nt < NT_TILES; ++nt) {
        v8f q = qacc[j][nt];
        v8h kh;
#pragma unroll
        for (int r = 0; r < 8; ++r) {
          // q path
          float vz = 0.5f * vq[j][nt][r] + (q[r] + biasq[j][r]);
          float sq = (vz >= 1.0f) ? 1.0f : 0.0f;
          vq[j][nt][r] = vz * (1.0f - sq);
          q[r] = sq;
          // k path -> packed f16 spike
          float vy = 0.5f * vk[j][nt][r] + (kacc[j][nt][r] + biask[j][r]);
          float sk = (vy >= 1.0f) ? 1.0f : 0.0f;
          vk[j][nt][r] = vy * (1.0f - sk);
          kh[r] = (_Float16)sk;
        }
        *(v8h*)&y_lds[nt * 16 + ln][mt * 16 + g * 8] = kh;   // ds_write_b128

        // temporal memory: m holds mems[t] already (updated at end of t-1)
        v8f m;
        if (t == 0) {
          m = q;                              // mems[0] = q[0]
        } else {
          m = mst[j][nt];
        }
        float part = 0.0f;
#pragma unroll
        for (int r = 0; r < 8; ++r) part += m[r] + q[r];
        atomicAdd(&attn_sum[h * NT + nt * 16 + ln], part);   // ds_add_f32
        // mems[t+1] = alpha*mems[t] + (1-alpha)*q[t]
#pragma unroll
        for (int r = 0; r < 8; ++r) m[r] = alpha * m[r] + (1.0f - alpha) * q[r];
        mst[j][nt] = m;
      }
    }
    __syncthreads();

    // ---- stage 4: attention LIF (threshold 0.5), one (head,col) per thread ----
    {
      const float va = 0.5f * v_attn + attn_sum[tid];
      const float sa = (va >= 0.5f) ? 1.0f : 0.0f;
      v_attn = va * (1.0f - sa);
      attn_spk[tid] = sa;
    }
    __syncthreads();

    // ---- stage 5: proj GEMM on (attn (*) k); attn gate folded into B-frag ----
    v8f pacc[MT_PER_WAVE][NT_TILES];
#pragma unroll
    for (int j = 0; j < MT_PER_WAVE; ++j)
#pragma unroll
      for (int nt = 0; nt < NT_TILES; ++nt) pacc[j][nt] = vzero;

#pragma unroll 1                     // rolled: bounds in-flight fragments
    for (int kt = 0; kt < KTILES; ++kt) {
      // each 16-wide K block (channels) lies inside one 48-channel head
      const int h = (kt * 32 + g * 16) / 48;
      Frag16 bf[NT_TILES];
#pragma unroll
      for (int nt = 0; nt < NT_TILES; ++nt) {
        const _Float16* yp = &y_lds[nt * 16 + ln][kt * 32 + g * 16];
        bf[nt].h[0] = *(const v8h*)yp;
        bf[nt].h[1] = *(const v8h*)(yp + 8);
        const _Float16 ah = (_Float16)attn_spk[h * NT + nt * 16 + ln];
#pragma unroll
        for (int i = 0; i < 16; ++i) bf[nt].v[i] *= ah;   // v_pk_mul_f16 x8
      }
#pragma unroll
      for (int j = 0; j < MT_PER_WAVE; ++j) {
        const int mt = wave + WAVES * j;
        const size_t row = (size_t)(mt * 16 + ln) * CCH + kt * 32 + g * 16;
        Frag16 ap;
        ap.h[0] = *(gv8_t)(wp_t + row);
        ap.h[1] = *(gv8_t)(wp_t + row + 8);
#pragma unroll
        for (int nt = 0; nt < NT_TILES; ++nt)
          pacc[j][nt] = __builtin_amdgcn_wmma_f32_16x16x32_f16(
              false, ap.v, false, bf[nt].v, (short)0, pacc[j][nt], false, false);
      }
    }

    // ---- stage 6: folded bias + store to HBM ----
    float* ot = out + ((size_t)(t * BATCH + b) * CCH) * NPIX + n0;
#pragma unroll
    for (int j = 0; j < MT_PER_WAVE; ++j) {
      const int mt = wave + WAVES * j;
#pragma unroll
      for (int nt = 0; nt < NT_TILES; ++nt)
#pragma unroll
        for (int r = 0; r < 8; ++r) {
          const int d = mt * 16 + g * 8 + r;
          ot[(size_t)d * NPIX + nt * 16 + ln] = pacc[j][nt][r] + biasp[j][r];
        }
    }
    __syncthreads();   // protect s_lds / y_lds before next timestep
  }
}

// ---------------------------------------------------------------------------
extern "C" void kernel_launch(void* const* d_in, const int* in_sizes, int n_in,
                              void* d_out, int out_size, void* d_ws, size_t ws_size,
                              hipStream_t stream)
{
  const float* x        = (const float*)d_in[0];
  const float* q_w      = (const float*)d_in[1];
  const float* q_gamma  = (const float*)d_in[2];
  const float* q_beta   = (const float*)d_in[3];
  const float* q_mean   = (const float*)d_in[4];
  const float* q_var    = (const float*)d_in[5];
  const float* k_w      = (const float*)d_in[6];
  const float* k_gamma  = (const float*)d_in[7];
  const float* k_beta   = (const float*)d_in[8];
  const float* k_mean   = (const float*)d_in[9];
  const float* k_var    = (const float*)d_in[10];
  const float* proj_w   = (const float*)d_in[11];
  const float* proj_b   = (const float*)d_in[12];
  const float* p_gamma  = (const float*)d_in[13];
  const float* p_beta   = (const float*)d_in[14];
  const float* p_mean   = (const float*)d_in[15];
  const float* p_var    = (const float*)d_in[16];
  const float* mem_alpha= (const float*)d_in[17];

  char* ws = (char*)d_ws;
  constexpr size_t WBYTES = (size_t)CCH * CCH * sizeof(_Float16);  // 294912
  _Float16* wqh = (_Float16*)(ws);
  _Float16* wkh = (_Float16*)(ws + WBYTES);
  _Float16* wph = (_Float16*)(ws + 2 * WBYTES);
  float* bq = (float*)(ws + 3 * WBYTES);
  float* bk = (float*)(ws + 3 * WBYTES + CCH * sizeof(float));
  float* bp = (float*)(ws + 3 * WBYTES + 2 * CCH * sizeof(float));

  const int prep_blocks = (CCH * CCH + 255) / 256;   // 576
  prep_weights<<<prep_blocks, 256, 0, stream>>>(q_w, q_gamma, q_beta, q_mean, q_var,
                                                nullptr, wqh, bq);
  prep_weights<<<prep_blocks, 256, 0, stream>>>(k_w, k_gamma, k_beta, k_mean, k_var,
                                                nullptr, wkh, bk);
  prep_weights<<<prep_blocks, 256, 0, stream>>>(proj_w, p_gamma, p_beta, p_mean, p_var,
                                                proj_b, wph, bp);

  const int blocks = BATCH * (NPIX / NT);            // 256 WGs x 256 threads
  snn_token_qk_attn<<<blocks, 256, 0, stream>>>(x, wqh, wkh, wph, bq, bk, bp,
                                                mem_alpha, (float*)d_out);
}
